// SimpleLSTM_69793218560433
// MI455X (gfx1250) — matrix operand order repaired and checked by reading, not compiled, not run on hardware
//
#include <hip/hip_runtime.h>
#include <hip/hip_bf16.h>

// LSTM: B=32, T=1024, F=512, H=512.
// Phase 1: Z = X @ Wi  (f32 WMMA, memory-bound, fully parallel) -> workspace
// Phase 2: persistent scan, 32 WGs x 256 thr; per-WG Wh slice resident in LDS
//          in fragment-ready bf16 layout (ds_load_b128 operand fetch);
//          v_wmma_f32_16x16x32_bf16 x16 per step; global_prefetch of the Zpre
//          slice; grid barrier (atomic arrive/wait) between timesteps.

#define B_   32
#define T_   1024
#define F_   512
#define H_   512
#define G4H  2048
#define NWG_SCAN 32      // = H_/16, each WG owns 16 H-columns
#define NH   16

typedef __attribute__((ext_vector_type(2)))  float  v2f;
typedef __attribute__((ext_vector_type(8)))  float  v8f;
typedef __attribute__((ext_vector_type(16))) __bf16 v16bf;

__global__ void init_ws(unsigned* counter) {
  if (threadIdx.x == 0) *counter = 0u;
}

// ---------------- Phase 1: Z[B*T,4H] = X[B*T,F] @ Wi[F,4H] ----------------
// Block computes a 64x32 tile; 8 waves, each one 16x16 WMMA tile.
__global__ __launch_bounds__(256)
void gemm_x_wi(const float* __restrict__ X, const float* __restrict__ Wi,
               float* __restrict__ Z) {
  __shared__ float As[64][33];   // 64 rows x 32 K (padded)
  __shared__ float Bs[32][33];   // 32 K x 32 cols (padded)

  const int tid  = threadIdx.x;
  const int w    = tid >> 5;
  const int lane = tid & 31;
  const int lm   = lane & 15;       // M/N index within tile
  const int lh   = lane >> 4;       // half-wave select
  const int m0   = (w & 3) * 16;    // 4 m-subtiles
  const int n0   = (w >> 2) * 16;   // 2 n-subtiles
  const int row0 = blockIdx.y * 64;
  const int col0 = blockIdx.x * 32;

  v8f acc = {};
  for (int kb = 0; kb < F_; kb += 32) {
    for (int i = tid; i < 64 * 32; i += 256)
      As[i >> 5][i & 31] = X[(long)(row0 + (i >> 5)) * F_ + kb + (i & 31)];
    for (int i = tid; i < 32 * 32; i += 256)
      Bs[i >> 5][i & 31] = Wi[(long)(kb + (i >> 5)) * G4H + col0 + (i & 31)];
    __syncthreads();
#pragma unroll
    for (int k = 0; k < 32; k += 4) {
      // f32 16x16x4 fragments: lanes 0-15 hold K={k,k+1}, lanes 16-31 K={k+2,k+3}
      v2f a, b;
      a.x = As[m0 + lm][k + 2 * lh + 0];
      a.y = As[m0 + lm][k + 2 * lh + 1];
      b.x = Bs[k + 2 * lh + 0][n0 + lm];
      b.y = Bs[k + 2 * lh + 1][n0 + lm];
      acc = __builtin_amdgcn_wmma_f32_16x16x4_f32(false, a, false, b,
                                                  (short)0, acc, false, false);
    }
    __syncthreads();
  }
  // C/D layout: VGPR j -> row j + 8*(lane/16), col lane%16
#pragma unroll
  for (int j = 0; j < 8; ++j)
    Z[(long)(row0 + m0 + j + 8 * lh) * G4H + col0 + n0 + lm] = acc[j];
}

// ---------------- Phase 2: persistent LSTM scan ----------------
// Fragment-ready LDS layouts (each lane's 16 bf16 elements contiguous, 32B):
//   wh_frag[kb5][gate][lane][e]  : B-frag for v_wmma_16x16x32_bf16
//   h_frag [kb5][rtile][lane][e] : A-frag
__global__ __launch_bounds__(256)
void lstm_scan(const float* __restrict__ Zpre, const float* __restrict__ h0,
               const float* __restrict__ c0, const float* __restrict__ Wh,
               const float* __restrict__ bias, float* __restrict__ out,
               unsigned* __restrict__ counter) {
  __shared__ __attribute__((aligned(32))) __bf16 wh_frag[16 * 4 * 32 * 16]; // 64 KB
  __shared__ __attribute__((aligned(32))) __bf16 h_frag [16 * 2 * 32 * 16]; // 32 KB
  __shared__ float z_s[8][16][16];   // 8 KB: per-wave gate tiles
  __shared__ float c_s[B_][NH];      // 2 KB: resident cell-state slice
  __shared__ float b_s[64];

  const int tid  = threadIdx.x;
  const int w    = tid >> 5;
  const int lane = tid & 31;
  const int lm   = lane & 15;
  const int lh   = lane >> 4;
  const int r    = w & 1;            // row tile (batch 0-15 / 16-31)
  const int g    = w >> 1;           // gate (i,f,g,o)
  const int hb   = blockIdx.x * NH;  // H-column base owned by this WG

  float* c_fin = out;                    // [B,H]
  float* h_fin = out + B_ * H_;          // [B,H]
  float* outs  = out + 2 * B_ * H_;      // [B,T,H]

  // One-time: park Wh slice in fragment order (bf16), bias + c0 slices.
  // B-frag mapping: K_local = 16*lh + e  ->  lh = (k%32)>>4, e = k&15
  for (int i = tid; i < F_ * 64; i += 256) {
    int k = i >> 6, c = i & 63;
    int gg = c >> 4, n = c & 15;
    int kb5 = k >> 5, kl = k & 31;
    int flh = kl >> 4, fe = kl & 15;
    int flane = flh * 16 + n;
    wh_frag[(((kb5 * 4 + gg) * 32) + flane) * 16 + fe] =
        (__bf16)Wh[(long)k * G4H + gg * H_ + hb + n];
  }
  for (int i = tid; i < 64; i += 256)
    b_s[i] = bias[(i >> 4) * H_ + hb + (i & 15)];
  for (int i = tid; i < B_ * NH; i += 256)
    c_s[i >> 4][i & 15] = c0[(i >> 4) * H_ + hb + (i & 15)];
  __syncthreads();

  for (int t = 0; t < T_; ++t) {
    // Broadcast h_{t-1} (32x512) into LDS as bf16 in A-fragment order.
    // A-frag mapping: K_local = 16*lh + e
    //   -> lh = kl>>4 ; e = kl&15
    for (int i = tid; i < B_ * H_; i += 256) {
      int bb = i >> 9, hh = i & 511;
      float hv = (t == 0) ? h0[i]
                          : outs[(long)bb * T_ * H_ + (long)(t - 1) * H_ + hh];
      int rr = bb >> 4, mm = bb & 15;
      int kb5 = hh >> 5, kl = hh & 31;
      int flh = kl >> 4;
      int fe  = kl & 15;
      int flane = flh * 16 + mm;
      h_frag[(((kb5 * 2 + rr) * 32) + flane) * 16 + fe] = (__bf16)hv;
    }
    __syncthreads();

    // Prefetch this step's Zpre slice while the WMMAs run.
#pragma unroll
    for (int j = 0; j < 8; ++j) {
      int brow = 16 * r + j + 8 * lh;
      __builtin_prefetch(&Zpre[((long)brow * T_ + t) * G4H + g * H_ + hb + lm], 0, 3);
    }

    // z_gate_tile = h @ Wh_slice  (K = 512, bf16 WMMA, fp32 accumulate)
    v8f acc = {};
#pragma unroll
    for (int kb5 = 0; kb5 < 16; ++kb5) {
      const v16bf a = *reinterpret_cast<const v16bf*>(
          &h_frag[(((kb5 * 2 + r) * 32) + lane) * 16]);
      const v16bf bf = *reinterpret_cast<const v16bf*>(
          &wh_frag[(((kb5 * 4 + g) * 32) + lane) * 16]);
      acc = __builtin_amdgcn_wmma_f32_16x16x32_bf16(false, a, false, bf,
                                                    (short)0, acc, false, false);
    }

    // Epilogue: z = acc + Zpre[:,t,:] + b  -> LDS gate tile
#pragma unroll
    for (int j = 0; j < 8; ++j) {
      int m    = j + 8 * lh;
      int brow = 16 * r + m;
      int col  = g * H_ + hb + lm;
      float z  = acc[j] + Zpre[((long)brow * T_ + t) * G4H + col] + b_s[g * 16 + lm];
      z_s[w][m][lm] = z;
    }
    __syncthreads();

    // Gate combine for this H-slice (512 elements).
    for (int e = tid; e < B_ * NH; e += 256) {
      int brow = e >> 4, hc = e & 15;
      int rr = brow >> 4, m = brow & 15;
      float zi = z_s[0 * 2 + rr][m][hc];
      float zf = z_s[1 * 2 + rr][m][hc];
      float zg = z_s[2 * 2 + rr][m][hc];
      float zo = z_s[3 * 2 + rr][m][hc];
      float ig = 1.f / (1.f + __expf(-zi));
      float fg = 1.f / (1.f + __expf(-zf));
      float gg = tanhf(zg);
      float og = 1.f / (1.f + __expf(-zo));
      float cn = fg * c_s[brow][hc] + ig * gg;
      float hn = og * tanhf(cn);
      c_s[brow][hc] = cn;
      outs[(long)brow * T_ * H_ + (long)t * H_ + hb + hc] = hn;
      if (t == T_ - 1) {
        c_fin[brow * H_ + hb + hc] = cn;
        h_fin[brow * H_ + hb + hc] = hn;
      }
    }

    // Grid-wide release/acquire barrier before next timestep.
    __threadfence();
    __syncthreads();
    if (tid == 0) {
      __hip_atomic_fetch_add(counter, 1u, __ATOMIC_ACQ_REL,
                             __HIP_MEMORY_SCOPE_AGENT);
      unsigned target = (unsigned)(NWG_SCAN * (t + 1));
      while (__hip_atomic_load(counter, __ATOMIC_ACQUIRE,
                               __HIP_MEMORY_SCOPE_AGENT) < target)
        __builtin_amdgcn_s_sleep(2);
    }
    __threadfence();
    __syncthreads();
  }
}

extern "C" void kernel_launch(void* const* d_in, const int* in_sizes, int n_in,
                              void* d_out, int out_size, void* d_ws, size_t ws_size,
                              hipStream_t stream) {
  (void)in_sizes; (void)n_in; (void)out_size; (void)ws_size;
  const float* x  = (const float*)d_in[0];
  const float* h0 = (const float*)d_in[1];
  const float* c0 = (const float*)d_in[2];
  const float* Wi = (const float*)d_in[3];
  const float* Wh = (const float*)d_in[4];
  const float* b  = (const float*)d_in[5];
  float* out = (float*)d_out;

  unsigned* counter = (unsigned*)d_ws;
  float* zpre = (float*)((char*)d_ws + 256);  // B*T*4H fp32 = 256 MB scratch

  init_ws<<<1, 32, 0, stream>>>(counter);
  gemm_x_wi<<<dim3(G4H / 32, (B_ * T_) / 64), 256, 0, stream>>>(x, Wi, zpre);
  lstm_scan<<<NWG_SCAN, 256, 0, stream>>>(zpre, h0, c0, Wh, b, out, counter);
}
